// Memory_9328668967233
// MI455X (gfx1250) — compile-verified, matrix operand order checked by reading, not run on hardware
//
#include <hip/hip_runtime.h>
#include <hip/hip_bf16.h>

// ---------------------------------------------------------------------------
// Memory-attention op for MI455X (gfx1250), wave32 + WMMA.
//   T=16 De=128 Do=512 Hm=Wm=32 -> N=16384 ; Bq=4 Hq=Wq=32 -> Q=1024
//   GEMM1: logits[b,n,q] = (miT[n,:] . qi[b,:,q]) / sqrt(128)
//   softmax over n ; GEMM2: mem[b,o,q] = mo[o,:] . p[b,:,q]
//   d_out = [ mem_out (4*1024*1024 f32) | p (4*16384*1024 f32) ]
// ---------------------------------------------------------------------------

typedef _Float16 half_t;
typedef __attribute__((ext_vector_type(8)))  _Float16 v8h;
typedef __attribute__((ext_vector_type(16))) _Float16 v16h;
typedef __attribute__((ext_vector_type(8)))  float    v8f;
typedef __attribute__((ext_vector_type(4)))  int      v4i;

#define GLOBAL_AS __attribute__((address_space(1)))
#define LDS_AS    __attribute__((address_space(3)))

#ifndef __has_builtin
#define __has_builtin(x) 0
#endif

#if defined(__HIP_DEVICE_COMPILE__)
#if __has_builtin(__builtin_amdgcn_global_load_async_to_lds_b128)
#define HAVE_ASYNC_LDS 1
#endif
#endif

static constexpr int   kN  = 16384;       // T*Hm*Wm
static constexpr int   kDe = 128;
static constexpr int   kDo = 512;
static constexpr int   kBq = 4;
static constexpr int   kQ  = 1024;
static constexpr int   kLdsStride = 40;   // halves per row (32 data + 8 pad), 80B (16B-aligned)

// ---- 16B global->LDS copy: async DMA if builtin exists, sync fallback otherwise
__device__ __forceinline__ void copy16_to_lds(half_t* lds_dst, const half_t* gsrc) {
#if defined(HAVE_ASYNC_LDS)
    __builtin_amdgcn_global_load_async_to_lds_b128(
        (GLOBAL_AS v4i*)gsrc, (LDS_AS v4i*)lds_dst, 0, 0);
#else
    *(v8h*)lds_dst = *(const v8h*)gsrc;
#endif
}

__device__ __forceinline__ void copy_wait() {
#if defined(HAVE_ASYNC_LDS)
#if __has_builtin(__builtin_amdgcn_s_wait_asynccnt)
    __builtin_amdgcn_s_wait_asynccnt(0);
#else
    asm volatile("s_wait_asynccnt 0" ::: "memory");
#endif
#endif
}

// ---- load one 16x32 f16 A/B fragment from an LDS tile stored row-major with
//      K contiguous (row stride kLdsStride halves). Lane mapping per CDNA5 ISA:
//      lanes 0-15: row = lane,  K halves {0..7, 16..23}
//      lanes16-31: row = lane-16, K halves {8..15, 24..31}
__device__ __forceinline__ v16h load_frag(const half_t* tile, int row, int koff) {
    const half_t* p0 = tile + row * kLdsStride + koff;
    v8h lo = *(const v8h*)(p0);
    v8h hi = *(const v8h*)(p0 + 16);
    v16h r;
#pragma unroll
    for (int i = 0; i < 8; ++i) { r[i] = lo[i]; r[i + 8] = hi[i]; }
    return r;
}

__device__ __forceinline__ v8f wmma_f16(v16h a, v16h b, v8f c) {
    return __builtin_amdgcn_wmma_f32_16x16x32_f16(false, a, false, b, (short)0, c,
                                                  false, false);
}

// ===========================================================================
// K0a: mi [128][16384] f32  ->  miT [16384][128] f16   (K-contiguous for WMMA A)
__global__ __launch_bounds__(256) void cvt_mi_kernel(const float* __restrict__ mi,
                                                     half_t* __restrict__ miT) {
    size_t idx = (size_t)blockIdx.x * 256 + threadIdx.x;   // 2,097,152
    int d = (int)(idx >> 14);
    int n = (int)(idx & (kN - 1));
    miT[(size_t)n * kDe + d] = (half_t)mi[idx];
}

// K0b: qi [4][128][1024] f32 -> qiT [4*1024][128] f16
__global__ __launch_bounds__(256) void cvt_qi_kernel(const float* __restrict__ qi,
                                                     half_t* __restrict__ qiT) {
    size_t idx = (size_t)blockIdx.x * 256 + threadIdx.x;   // 524,288
    int b = (int)(idx >> 17);
    int rem = (int)(idx & 131071);
    int d = rem >> 10;
    int q = rem & (kQ - 1);
    qiT[((size_t)b * kQ + q) * kDe + d] = (half_t)qi[idx];
}

// K0c: mo [512][16384] f32 -> moH same layout f16 (already K-contiguous)
__global__ __launch_bounds__(256) void cvt_mo_kernel(const float* __restrict__ mo,
                                                     half_t* __restrict__ moH) {
    size_t i4 = (size_t)blockIdx.x * 256 + threadIdx.x;    // 2,097,152 float4s
    float4 v = *(const float4*)(mo + i4 * 4);
    half_t* o = moH + i4 * 4;
    o[0] = (half_t)v.x; o[1] = (half_t)v.y; o[2] = (half_t)v.z; o[3] = (half_t)v.w;
}

// ===========================================================================
// K1: GEMM1  logits[b,n,q] = scale * sum_d miT[n,d]*qiT[b*Q+q,d]
//     tile: 128n x 64q per WG, K loop 4 x 32. 8 waves; wave w owns n-rows w*16..+15.
__global__ __launch_bounds__(256) void gemm1_kernel(const half_t* __restrict__ miT,
                                                    const half_t* __restrict__ qiT,
                                                    float* __restrict__ logits) {
    __shared__ __attribute__((aligned(16))) half_t At[128 * kLdsStride];
    __shared__ __attribute__((aligned(16))) half_t Bt[64 * kLdsStride];

    const int qBase = blockIdx.x * 64;
    const int nBase = blockIdx.y * 128;
    const int b     = blockIdx.z;
    const int t     = threadIdx.x;
    const int w     = t >> 5;
    const int lane  = t & 31;
    const int m     = lane & 15;
    const int koff  = (lane >> 4) * 8;

    v8f acc[4];
#pragma unroll
    for (int i = 0; i < 4; ++i)
        acc[i] = (v8f){0.f, 0.f, 0.f, 0.f, 0.f, 0.f, 0.f, 0.f};

    for (int k0 = 0; k0 < kDe; k0 += 32) {
        // A tile: 128 rows x 4 chunks = 512 chunks, 2 per thread
#pragma unroll
        for (int i = 0; i < 2; ++i) {
            int cid = i * 256 + t;
            int row = cid >> 2, cs = cid & 3;
            copy16_to_lds(&At[row * kLdsStride + cs * 8],
                          &miT[(size_t)(nBase + row) * kDe + k0 + cs * 8]);
        }
        // B tile: 64 rows x 4 chunks = 256 chunks, 1 per thread
        {
            int row = t >> 2, cs = t & 3;
            copy16_to_lds(&Bt[row * kLdsStride + cs * 8],
                          &qiT[((size_t)b * kQ + qBase + row) * kDe + k0 + cs * 8]);
        }
        copy_wait();
        __syncthreads();

        v16h a = load_frag(At, w * 16 + m, koff);
#pragma unroll
        for (int qs = 0; qs < 4; ++qs) {
            v16h bb = load_frag(Bt, qs * 16 + m, koff);
            acc[qs] = wmma_f16(a, bb, acc[qs]);
        }
        __syncthreads();
    }

    const float scale = 0.08838834764831845f;   // 1/sqrt(128)
#pragma unroll
    for (int qs = 0; qs < 4; ++qs) {
        int q = qBase + qs * 16 + m;
#pragma unroll
        for (int r = 0; r < 8; ++r) {
            int n = nBase + w * 16 + r + ((lane >> 4) * 8);
            logits[((size_t)b * kN + n) * kQ + q] = acc[qs][r] * scale;
        }
    }
}

// ===========================================================================
// K2: per-(b, 256-row n-block, q) online max/sumexp partials
__global__ __launch_bounds__(256) void softmax_partial_kernel(
        const float* __restrict__ logits, float2* __restrict__ partial) {
    int q  = blockIdx.x * 256 + threadIdx.x;
    int nb = blockIdx.y;            // 0..63, 256 n-rows each
    int b  = blockIdx.z;
    const float* p = logits + ((size_t)b * kN + (size_t)nb * 256) * kQ + q;
    float m = -__builtin_inff(), s = 0.f;
    for (int i = 0; i < 256; ++i) {
        float x  = p[(size_t)i * kQ];
        float mn = fmaxf(m, x);
        s = s * __expf(m - mn) + __expf(x - mn);
        m = mn;
    }
    partial[((size_t)b * 64 + nb) * kQ + q] = make_float2(m, s);
}

// K3: combine 64 partials -> stats[b][q] = (max, 1/sum)
__global__ __launch_bounds__(256) void softmax_combine_kernel(
        const float2* __restrict__ partial, float2* __restrict__ stats) {
    int q = blockIdx.x * 256 + threadIdx.x;
    int b = blockIdx.y;
    const float2* pp = partial + (size_t)b * 64 * kQ + q;
    float M = -__builtin_inff(), S = 0.f;
    for (int i = 0; i < 64; ++i) {
        float2 v  = pp[(size_t)i * kQ];
        float  Mn = fmaxf(M, v.x);
        S = S * __expf(M - Mn) + v.y * __expf(v.x - Mn);
        M = Mn;
    }
    stats[(size_t)b * kQ + q] = make_float2(M, 1.0f / S);
}

// K4: normalize p in place: p = exp(x - M) * (1/S), float4 per thread
__global__ __launch_bounds__(256) void softmax_normalize_kernel(
        float* __restrict__ logits, const float2* __restrict__ stats) {
    size_t idx = ((size_t)blockIdx.x * 256 + threadIdx.x) * 4;  // element index
    int q = (int)(idx & (kQ - 1));
    int b = (int)(idx >> 24);                                   // N*Q = 2^24
    float4 v = *(const float4*)(logits + idx);
    const float2* st = stats + (size_t)b * kQ + q;
    float2 s0 = st[0], s1 = st[1], s2 = st[2], s3 = st[3];
    v.x = __expf(v.x - s0.x) * s0.y;
    v.y = __expf(v.y - s1.x) * s1.y;
    v.z = __expf(v.z - s2.x) * s2.y;
    v.w = __expf(v.w - s3.x) * s3.y;
    *(float4*)(logits + idx) = v;
}

// ===========================================================================
// K5: GEMM2  mem[b,o,q] = sum_n moH[o,n] * p[b,n,q]
//     WG tile: full o=512 x 32q (p read exactly once), K loop 512 x 32.
//     8 waves; wave w owns o-rows w*64..w*64+63 (4 o-subtiles x 2 q-subtiles).
__global__ __launch_bounds__(256) void gemm2_kernel(const half_t* __restrict__ moH,
                                                    const float* __restrict__ p,
                                                    float* __restrict__ out) {
    __shared__ __attribute__((aligned(16))) half_t At[512 * kLdsStride];  // 40 KB
    __shared__ __attribute__((aligned(16))) half_t Bt[32 * kLdsStride];

    const int qBase = blockIdx.x * 32;
    const int b     = blockIdx.y;
    const int t     = threadIdx.x;
    const int w     = t >> 5;
    const int lane  = t & 31;
    const int m     = lane & 15;
    const int koff  = (lane >> 4) * 8;

    v8f acc[4][2];
#pragma unroll
    for (int i = 0; i < 4; ++i)
#pragma unroll
        for (int j = 0; j < 2; ++j)
            acc[i][j] = (v8f){0.f, 0.f, 0.f, 0.f, 0.f, 0.f, 0.f, 0.f};

    const float* pB = p + (size_t)b * kN * kQ + qBase;

    for (int n0 = 0; n0 < kN; n0 += 32) {
        // A tile: 512 rows x 4 chunks = 2048 chunks, 8 per thread (async DMA)
#pragma unroll
        for (int i = 0; i < 8; ++i) {
            int cid = i * 256 + t;
            int row = cid >> 2, cs = cid & 3;
            copy16_to_lds(&At[row * kLdsStride + cs * 8],
                          &moH[(size_t)row * kN + n0 + cs * 8]);
        }
        // B tile: p[n0..n0+31][qBase..+31] f32 -> f16 transposed into [q][k]
#pragma unroll
        for (int i = 0; i < 4; ++i) {
            int id = i * 256 + t;
            int kk = id >> 5, qq = id & 31;
            Bt[qq * kLdsStride + kk] = (half_t)pB[(size_t)(n0 + kk) * kQ + qq];
        }
        copy_wait();
        __syncthreads();

        v16h b0 = load_frag(Bt, m, koff);
        v16h b1 = load_frag(Bt, 16 + m, koff);
#pragma unroll
        for (int os = 0; os < 4; ++os) {
            v16h a = load_frag(At, w * 64 + os * 16 + m, koff);
            acc[os][0] = wmma_f16(a, b0, acc[os][0]);
            acc[os][1] = wmma_f16(a, b1, acc[os][1]);
        }
        __syncthreads();
    }

    // mem_out[b][o][q], channel dim is Do*2=1024 with mem in channels [0,512)
#pragma unroll
    for (int os = 0; os < 4; ++os) {
#pragma unroll
        for (int qs = 0; qs < 2; ++qs) {
            int q = qBase + qs * 16 + m;
#pragma unroll
            for (int r = 0; r < 8; ++r) {
                int o = w * 64 + os * 16 + r + ((lane >> 4) * 8);
                out[(size_t)b * (1024 * 1024) + (size_t)o * 1024 + q] = acc[os][qs][r];
            }
        }
    }
}

// K6: q_out [4][512][1024] -> mem_out channels [512,1024)
__global__ __launch_bounds__(256) void copy_qout_kernel(const float* __restrict__ qo,
                                                        float* __restrict__ out) {
    size_t i4 = (size_t)blockIdx.x * 256 + threadIdx.x;   // 524,288 float4s
    size_t e  = i4 * 4;
    int b = (int)(e >> 19);                               // 512*1024 per batch
    size_t rem = e & ((1u << 19) - 1);
    float4 v = *(const float4*)(qo + e);
    *(float4*)(out + (size_t)b * (1u << 20) + (1u << 19) + rem) = v;
}

// ===========================================================================
extern "C" void kernel_launch(void* const* d_in, const int* in_sizes, int n_in,
                              void* d_out, int out_size, void* d_ws, size_t ws_size,
                              hipStream_t stream) {
    (void)in_sizes; (void)n_in; (void)out_size; (void)ws_size;

    const float* m_in  = (const float*)d_in[0];   // [16][128][32][32]
    const float* m_out = (const float*)d_in[1];   // [16][512][32][32]
    const float* q_in  = (const float*)d_in[2];   // [4][128][32][32]
    const float* q_out = (const float*)d_in[3];   // [4][512][32][32]

    float* out    = (float*)d_out;                // mem_out: 4*1024*1024
    float* logits = out + 4194304;                // p: 4*16384*1024

    char*   ws    = (char*)d_ws;                  // ~23.1 MB used
    half_t* miT   = (half_t*)(ws);                //  4 MB   [16384][128]
    half_t* qiT   = (half_t*)(ws + 4194304);      //  1 MB   [4096][128]
    half_t* moH   = (half_t*)(ws + 5242880);      // 16 MB   [512][16384]
    float2* part  = (float2*)(ws + 22020096);     //  2 MB   [4][64][1024]
    float2* stats = (float2*)(ws + 24117248);     // 32 KB   [4][1024]

    cvt_mi_kernel<<<8192, 256, 0, stream>>>(m_in, miT);
    cvt_qi_kernel<<<2048, 256, 0, stream>>>(q_in, qiT);
    cvt_mo_kernel<<<8192, 256, 0, stream>>>(m_out, moH);

    gemm1_kernel<<<dim3(kQ / 64, kN / 128, kBq), 256, 0, stream>>>(miT, qiT, logits);

    softmax_partial_kernel<<<dim3(kQ / 256, 64, kBq), 256, 0, stream>>>(logits, part);
    softmax_combine_kernel<<<dim3(kQ / 256, kBq), 256, 0, stream>>>(part, stats);
    softmax_normalize_kernel<<<65536, 256, 0, stream>>>(logits, stats);

    gemm2_kernel<<<dim3(kQ / 32, kBq), 256, 0, stream>>>(moH, logits, out);

    copy_qout_kernel<<<2048, 256, 0, stream>>>(q_out, out);
}
